// MambaDeter_26903675142799
// MI455X (gfx1250) — matrix operand order repaired and checked by reading, not compiled, not run on hardware
//
#include <hip/hip_runtime.h>
#include <math.h>

// ---------------------------------------------------------------- constants
#define B_SZ     1024
#define STOCH    1024
#define ACT      32
#define DETER    4096
#define TOKEN_D  512
#define HIDDEN   1024
#define NHEADS   8
#define HEADDIM  128
#define DSTATE   64
#define CONV_D   1152           // HIDDEN + 2*DSTATE
#define PROJ_D   2184           // HIDDEN + CONV_D + NHEADS
#define INP_D    1056           // STOCH + ACT
#define EPSV     1e-4f

typedef __attribute__((ext_vector_type(16))) __bf16 v16bf;
typedef __attribute__((ext_vector_type(8)))  float  v8f;

__device__ __forceinline__ float silu_f(float x) { return x / (1.f + __expf(-x)); }

__device__ __forceinline__ v16bf cvt_frag(float4 a, float4 b, float4 c, float4 d) {
  v16bf r;
  r[0]  = (__bf16)a.x; r[1]  = (__bf16)a.y; r[2]  = (__bf16)a.z; r[3]  = (__bf16)a.w;
  r[4]  = (__bf16)b.x; r[5]  = (__bf16)b.y; r[6]  = (__bf16)b.z; r[7]  = (__bf16)b.w;
  r[8]  = (__bf16)c.x; r[9]  = (__bf16)c.y; r[10] = (__bf16)c.z; r[11] = (__bf16)c.w;
  r[12] = (__bf16)d.x; r[13] = (__bf16)d.y; r[14] = (__bf16)d.z; r[15] = (__bf16)d.w;
  return r;
}
// A fragment, p already points at (row, k0 + hiK): elems 0..7 = p[0..7], 8..15 = p[16..23]
__device__ __forceinline__ v16bf load_a_frag(const float* __restrict__ p) {
  const float4* q0 = (const float4*)p;
  const float4* q1 = (const float4*)(p + 16);
  return cvt_frag(q0[0], q0[1], q1[0], q1[1]);
}
// B fragment, p points at (col, k0 + kOffB): elems 0..7 = p[0..7], 8..15 = p[16..23]
__device__ __forceinline__ v16bf load_b_frag(const float* __restrict__ p) {
  const float4* q0 = (const float4*)p;
  const float4* q1 = (const float4*)(p + 16);
  return cvt_frag(q0[0], q0[1], q1[0], q1[1]);
}
__device__ __forceinline__ v8f wmma_bf16(v16bf a, v16bf b, v8f c) {
  return __builtin_amdgcn_wmma_f32_16x16x32_bf16(false, a, false, b, (short)0, c,
                                                 false, false);
}

// ---------------------------------------------------------------- kernel 0
__global__ void build_inp_kernel(const float* __restrict__ stoch,
                                 const float* __restrict__ action,
                                 float* __restrict__ inp) {
  int i = blockIdx.x * blockDim.x + threadIdx.x;
  if (i >= B_SZ * INP_D) return;
  int b = i / INP_D, c = i - b * INP_D;
  float v;
  if (c < STOCH) {
    v = stoch[b * STOCH + c];
  } else {
    float a = action[b * ACT + (c - STOCH)];
    float d = fabsf(a);
    v = a / (d > 1.f ? d : 1.f);
  }
  inp[i] = v;
}

// ---------------------------------------------------------------- WMMA GEMM
// out[M,N] = A[M,K] @ W[N,K]^T (+bias) (+residual)
// One wave computes a 32x32 output block as a 2x2 grid of 16x16 WMMA tiles:
// 4 WMMAs per K-step, each A/B fragment reused twice.
// K multiple of 32; M multiple of 32; N arbitrary (clamped loads, masked stores).
template <bool HAS_BIAS, bool HAS_RES>
__global__ void wmma_gemm_kernel(const float* __restrict__ A,
                                 const float* __restrict__ W,
                                 const float* __restrict__ bias,
                                 const float* __restrict__ residual,
                                 float* __restrict__ out,
                                 int M, int N, int K) {
  const int lane    = threadIdx.x & 31;
  const int blocksN = (N + 31) >> 5;
  const int blocksM = M >> 5;
  const int gwave   = (blockIdx.x << 3) + (threadIdx.x >> 5);
  if (gwave >= blocksM * blocksN) return;          // wave-uniform exit
  const int bm = gwave / blocksN;
  const int bn = gwave - bm * blocksN;

  const int l15 = lane & 15;
  const int lhi = lane >> 4;                       // 0 or 1
  const int m0  = bm * 32 + l15;                   // A rows owned by this lane
  const int m1  = m0 + 16;
  const int n0r = bn * 32 + l15;                   // D columns owned by this lane
  const int n1r = n0r + 16;
  const int n0  = n0r < N ? n0r : N - 1;           // clamped for loads
  const int n1  = n1r < N ? n1r : N - 1;
  const int hiK   = lhi * 8;
  const int kOffB = lhi * 8;

  const float* __restrict__ A0 = A + (size_t)m0 * K + hiK;
  const float* __restrict__ A1 = A + (size_t)m1 * K + hiK;
  const float* __restrict__ W0 = W + (size_t)n0 * K + kOffB;
  const float* __restrict__ W1 = W + (size_t)n1 * K + kOffB;

  v8f acc00 = {}, acc01 = {}, acc10 = {}, acc11 = {};
  for (int k0 = 0; k0 < K; k0 += 32) {
    v16bf a0 = load_a_frag(A0 + k0);
    v16bf a1 = load_a_frag(A1 + k0);
    v16bf b0 = load_b_frag(W0 + k0);
    v16bf b1 = load_b_frag(W1 + k0);
    acc00 = wmma_bf16(a0, b0, acc00);
    acc01 = wmma_bf16(a0, b1, acc01);
    acc10 = wmma_bf16(a1, b0, acc10);
    acc11 = wmma_bf16(a1, b1, acc11);
  }

  auto store_tile = [&](v8f acc, int mbase, int ncol) {
    if (ncol < N) {
      float bv = HAS_BIAS ? bias[ncol] : 0.f;
#pragma unroll
      for (int r = 0; r < 8; ++r) {
        size_t idx = (size_t)(mbase + r) * N + ncol;
        float v = acc[r] + bv;
        if (HAS_RES) v += residual[idx];
        out[idx] = v;
      }
    }
  };
  const int mb = bm * 32 + lhi * 8;
  store_tile(acc00, mb,      n0r);
  store_tile(acc01, mb,      n1r);
  store_tile(acc10, mb + 16, n0r);
  store_tile(acc11, mb + 16, n1r);
}

// ---------------------------------------------------------------- rowwise rmsnorm (512 cols)
__global__ void rmsnorm512_kernel(const float* __restrict__ in,
                                  const float* __restrict__ w,
                                  float* __restrict__ out, int apply_silu) {
  __shared__ float red[256];
  const int b = blockIdx.x, tid = threadIdx.x;
  const float* row = in + (size_t)b * TOKEN_D;
  float v0 = row[tid], v1 = row[tid + 256];
  red[tid] = v0 * v0 + v1 * v1;
  __syncthreads();
  for (int s = 128; s > 0; s >>= 1) {
    if (tid < s) red[tid] += red[tid + s];
    __syncthreads();
  }
  const float scale = rsqrtf(red[0] * (1.f / TOKEN_D) + EPSV);
  float o0 = v0 * scale * w[tid];
  float o1 = v1 * scale * w[tid + 256];
  if (apply_silu) { o0 = silu_f(o0); o1 = silu_f(o1); }
  float* orow = out + (size_t)b * TOKEN_D;
  orow[tid] = o0;
  orow[tid + 256] = o1;
}

// ---------------------------------------------------------------- fused conv + SSM + gate
// One workgroup (256 threads) per batch element. Streams the 256 KB of
// ssm_state for this row exactly once (read + write) via float4.
__global__ void conv_ssm_gate_kernel(const float* __restrict__ proj,
                                     const float* __restrict__ cs_in,
                                     const float* __restrict__ ss_in,
                                     const float* __restrict__ cw,
                                     const float* __restrict__ cb,
                                     const float* __restrict__ alog,
                                     const float* __restrict__ Dsk,
                                     const float* __restrict__ dtb,
                                     const float* __restrict__ pw,
                                     float* __restrict__ cs_out,
                                     float* __restrict__ ss_out,
                                     float* __restrict__ gated) {
  const int b = blockIdx.x, tid = threadIdx.x;
  __shared__ __attribute__((aligned(16))) float s_conv[CONV_D];
  __shared__ float s_y[HIDDEN];
  __shared__ float s_dt[NHEADS], s_dA[NHEADS];
  __shared__ float s_red[256];

  const float* projb = proj + (size_t)b * PROJ_D;

  if (tid < NHEADS) {
    float draw = projb[HIDDEN + CONV_D + tid] + dtb[tid];
    float dtv = (draw > 20.f) ? draw : log1pf(__expf(draw));   // softplus
    s_dt[tid] = dtv;
    s_dA[tid] = __expf(-__expf(alog[tid]) * dtv);
  }
  // depthwise K=4 conv over [cs0, cs1, cs2, xBC] + state shift-out
  for (int c = tid; c < CONV_D; c += 256) {
    size_t base = ((size_t)b * CONV_D + c) * 3;
    float x0 = cs_in[base], x1 = cs_in[base + 1], x2 = cs_in[base + 2];
    float x3 = projb[HIDDEN + c];
    float4 wv = *(const float4*)(cw + c * 4);
    float acc = wv.x * x0 + wv.y * x1 + wv.z * x2 + wv.w * x3 + cb[c];
    s_conv[c] = silu_f(acc);
    cs_out[base] = x1; cs_out[base + 1] = x2; cs_out[base + 2] = x3;
  }
  __syncthreads();

  // SSM: thread owns 4 full (h,p) rows of 64 states each (contiguous 1 KB)
  const float* ssb  = ss_in  + (size_t)b * (NHEADS * HEADDIM * DSTATE);
  float*       nssb = ss_out + (size_t)b * (NHEADS * HEADDIM * DSTATE);
#pragma unroll
  for (int r = 0; r < 4; ++r) {
    const int hp = tid * 4 + r;
    const int hh = hp >> 7;
    const float xv = s_conv[hp];
    const float dAv = s_dA[hh];
    const float coef = s_dt[hh] * xv;
    const float4* sp = (const float4*)(ssb + (size_t)hp * DSTATE);
    float4*       np = (float4*)(nssb + (size_t)hp * DSTATE);
    float yacc = 0.f;
#pragma unroll
    for (int i = 0; i < DSTATE / 4; ++i) {
      float4 s  = sp[i];
      float4 Bv = *(const float4*)(s_conv + HIDDEN + i * 4);
      float4 Cv = *(const float4*)(s_conv + HIDDEN + DSTATE + i * 4);
      float4 ns;
      ns.x = s.x * dAv + coef * Bv.x;
      ns.y = s.y * dAv + coef * Bv.y;
      ns.z = s.z * dAv + coef * Bv.z;
      ns.w = s.w * dAv + coef * Bv.w;
      np[i] = ns;
      yacc += ns.x * Cv.x + ns.y * Cv.y + ns.z * Cv.z + ns.w * Cv.w;
    }
    s_y[hp] = yacc + Dsk[hh] * xv;
  }
  __syncthreads();

  // gated = rmsnorm(y * silu(z), pw)
  float vals[4], ssum = 0.f;
#pragma unroll
  for (int j = 0; j < 4; ++j) {
    int c = tid + j * 256;
    float z = projb[c];
    float g = s_y[c] * silu_f(z);
    vals[j] = g;
    ssum += g * g;
  }
  s_red[tid] = ssum;
  __syncthreads();
  for (int s = 128; s > 0; s >>= 1) {
    if (tid < s) s_red[tid] += s_red[tid + s];
    __syncthreads();
  }
  const float scale = rsqrtf(s_red[0] * (1.f / HIDDEN) + EPSV);
  float* gb = gated + (size_t)b * HIDDEN;
#pragma unroll
  for (int j = 0; j < 4; ++j) {
    int c = tid + j * 256;
    gb[c] = vals[j] * scale * pw[c];
  }
}

// ---------------------------------------------------------------- final: rmsnorm + tape shift
__global__ void final_deter_kernel(const float* __restrict__ x,
                                   const float* __restrict__ w,
                                   const float* __restrict__ deter,
                                   float* __restrict__ deter_out) {
  __shared__ float red[256];
  const int b = blockIdx.x, tid = threadIdx.x;
  const float* row = x + (size_t)b * TOKEN_D;
  float v0 = row[tid], v1 = row[tid + 256];
  red[tid] = v0 * v0 + v1 * v1;
  __syncthreads();
  for (int s = 128; s > 0; s >>= 1) {
    if (tid < s) red[tid] += red[tid + s];
    __syncthreads();
  }
  const float scale = rsqrtf(red[0] * (1.f / TOKEN_D) + EPSV);
  float*       ob = deter_out + (size_t)b * DETER;
  const float* ib = deter     + (size_t)b * DETER;
  for (int i = tid; i < DETER - TOKEN_D; i += 256) ob[i] = ib[i + TOKEN_D];
  ob[DETER - TOKEN_D + tid]       = v0 * scale * w[tid];
  ob[DETER - TOKEN_D + tid + 256] = v1 * scale * w[tid + 256];
}

// ---------------------------------------------------------------- host
static void launch_gemm(const float* A, const float* W, const float* bias,
                        const float* residual, float* out,
                        int M, int N, int K, hipStream_t s) {
  int blocksM = M >> 5, blocksN = (N + 31) >> 5;
  int waves = blocksM * blocksN;
  int blocks = (waves + 7) / 8;                 // 8 waves (256 thr) per block
  if (bias && residual)
    wmma_gemm_kernel<true, true><<<blocks, 256, 0, s>>>(A, W, bias, residual, out, M, N, K);
  else if (bias)
    wmma_gemm_kernel<true, false><<<blocks, 256, 0, s>>>(A, W, bias, residual, out, M, N, K);
  else if (residual)
    wmma_gemm_kernel<false, true><<<blocks, 256, 0, s>>>(A, W, bias, residual, out, M, N, K);
  else
    wmma_gemm_kernel<false, false><<<blocks, 256, 0, s>>>(A, W, bias, residual, out, M, N, K);
}

extern "C" void kernel_launch(void* const* d_in, const int* in_sizes, int n_in,
                              void* d_out, int out_size, void* d_ws, size_t ws_size,
                              hipStream_t stream) {
  (void)in_sizes; (void)n_in; (void)out_size; (void)ws_size;
  const float* stoch       = (const float*)d_in[0];
  const float* deter       = (const float*)d_in[1];
  const float* action      = (const float*)d_in[2];
  const float* conv_state  = (const float*)d_in[3];
  const float* ssm_state   = (const float*)d_in[4];
  const float* in_proj_w   = (const float*)d_in[5];
  const float* in_proj_b   = (const float*)d_in[6];
  const float* in_norm_w   = (const float*)d_in[7];
  const float* blk_norm_w  = (const float*)d_in[8];
  const float* blk_in_w    = (const float*)d_in[9];
  const float* blk_conv_w  = (const float*)d_in[10];
  const float* blk_conv_b  = (const float*)d_in[11];
  const float* blk_A_log   = (const float*)d_in[12];
  const float* blk_D       = (const float*)d_in[13];
  const float* blk_dt_bias = (const float*)d_in[14];
  const float* blk_post_w  = (const float*)d_in[15];
  const float* blk_out_w   = (const float*)d_in[16];
  const float* blk_out_b   = (const float*)d_in[17];
  const float* out_norm_w  = (const float*)d_in[18];

  // output layout: [new_deter | new_conv_state | new_ssm_state]
  float* outp      = (float*)d_out;
  float* deter_out = outp;
  float* conv_out  = outp + (size_t)B_SZ * DETER;
  float* ssm_out   = conv_out + (size_t)2 * B_SZ * CONV_D * 3;

  // workspace layout
  float* ws    = (float*)d_ws;
  float* inp   = ws;                                   // 1024*1056
  float* x     = inp  + (size_t)B_SZ * INP_D;          // 1024*512
  float* h     = x    + (size_t)B_SZ * TOKEN_D;        // 1024*512
  float* proj  = h    + (size_t)B_SZ * TOKEN_D;        // 1024*2184
  float* gated = proj + (size_t)B_SZ * PROJ_D;         // 1024*1024

  // 1) inp = concat(stoch, action/max(|action|,1))
  {
    int total = B_SZ * INP_D;
    build_inp_kernel<<<(total + 255) / 256, 256, 0, stream>>>(stoch, action, inp);
  }
  // 2) tok = silu(rmsnorm(inp @ in_proj_w^T + b))   (raw into h, normed into x)
  launch_gemm(inp, in_proj_w, in_proj_b, nullptr, h, B_SZ, TOKEN_D, INP_D, stream);
  rmsnorm512_kernel<<<B_SZ, 256, 0, stream>>>(h, in_norm_w, x, /*silu=*/1);

  // 3) two mamba blocks
  for (int l = 0; l < 2; ++l) {
    const float* cs_l = conv_state + (size_t)l * B_SZ * CONV_D * 3;
    const float* ss_l = ssm_state  + (size_t)l * B_SZ * NHEADS * HEADDIM * DSTATE;
    float* cs_o = conv_out + (size_t)l * B_SZ * CONV_D * 3;
    float* ss_o = ssm_out  + (size_t)l * B_SZ * NHEADS * HEADDIM * DSTATE;

    rmsnorm512_kernel<<<B_SZ, 256, 0, stream>>>(x, blk_norm_w + l * TOKEN_D, h, /*silu=*/0);
    launch_gemm(h, blk_in_w + (size_t)l * PROJ_D * TOKEN_D, nullptr, nullptr,
                proj, B_SZ, PROJ_D, TOKEN_D, stream);
    conv_ssm_gate_kernel<<<B_SZ, 256, 0, stream>>>(
        proj, cs_l, ss_l,
        blk_conv_w + (size_t)l * CONV_D * 4, blk_conv_b + (size_t)l * CONV_D,
        blk_A_log + l * NHEADS, blk_D + l * NHEADS, blk_dt_bias + l * NHEADS,
        blk_post_w + (size_t)l * HIDDEN,
        cs_o, ss_o, gated);
    launch_gemm(gated, blk_out_w + (size_t)l * TOKEN_D * HIDDEN,
                blk_out_b + l * TOKEN_D, /*residual=*/x, x,
                B_SZ, TOKEN_D, HIDDEN, stream);
  }

  // 4) tape shift + final rmsnorm token
  final_deter_kernel<<<B_SZ, 256, 0, stream>>>(x, out_norm_w, deter, deter_out);
}
